// HybridCongestionModel_2138893714289
// MI455X (gfx1250) — matrix operand order repaired and checked by reading, not compiled, not run on hardware
//
#include <hip/hip_runtime.h>
#include <hip/hip_bf16.h>

// ---------------------------------------------------------------------------
// CDNA5 (gfx1250) GCN hybrid-congestion model.
// Dense transforms use v_wmma_f32_16x16x32_bf16 (wave32 WMMA, f32 accum).
// Edge aggregation uses hardware f32 atomics (L2-resident: xw fits in 192MB L2).
// ---------------------------------------------------------------------------

typedef __attribute__((ext_vector_type(16))) __bf16 v16bf;
typedef __attribute__((ext_vector_type(8)))  float  v8f;

static __device__ __forceinline__ __bf16 f2bf(float f) { return (__bf16)f; }

// ---------------------------------------------------------------------------
// Utility fills
// ---------------------------------------------------------------------------
__global__ void k_fill(float* __restrict__ p, int count, float v) {
  int i = blockIdx.x * blockDim.x + threadIdx.x;
  if (i < count) p[i] = v;
}

// deg[dst[e]] += 1   (deg pre-filled with 1.0 for the self loop)
__global__ void k_count_deg(const int* __restrict__ dst, float* __restrict__ deg, int ne) {
  int e = blockIdx.x * blockDim.x + threadIdx.x;
  if (e < ne) unsafeAtomicAdd(&deg[dst[e]], 1.0f);
}

// dinv = rsqrt(deg)   (deg >= 1 always: self loops)
__global__ void k_rsqrt(float* __restrict__ d, int n) {
  int i = blockIdx.x * blockDim.x + threadIdx.x;
  if (i < n) d[i] = rsqrtf(d[i]);
}

// ---------------------------------------------------------------------------
// GEMM: C[n x 64] = A[n x K] @ W[K x 64]   (f32 in memory, bf16 WMMA, f32 acc)
// One wave computes a 16x64 tile: 4 N-tiles x (K/32) WMMA ops.
// W is staged in LDS pre-swizzled into B fragments in this code's contiguous k order
// (lane<16: K=kb*32+0..15 elems, col=lane; lane>=16: K=kb*32+16..31, col=lane-16).
// A fragment k order used here (contiguous, same as B): lane<16 row=lane elems {K0..15},
// lane>=16 row=lane-16 elems {K16..31}.
// ---------------------------------------------------------------------------
#define GEMM_WAVES 8

template <int K>
__global__ void k_gemm_bf16(const float* __restrict__ A, const float* __restrict__ W,
                            float* __restrict__ C, int n) {
  constexpr int NKB = K / 32;
  __shared__ v16bf ldsB[NKB][4][32];

  const int tid = threadIdx.x;

  // Cooperative fill of B fragments (W is tiny: K*64 elements, each used once).
  for (int idx = tid; idx < NKB * 4 * 32; idx += blockDim.x) {
    const int kb   = idx >> 7;        // 4*32 = 128 entries per kb
    const int rem  = idx & 127;
    const int nt   = rem >> 5;
    const int ln   = rem & 31;
    const int col  = nt * 16 + (ln & 15);
    const int kbas = kb * 32 + (ln >> 4) * 16;
    v16bf frag;
#pragma unroll
    for (int i = 0; i < 16; ++i) frag[i] = f2bf(W[(kbas + i) * 64 + col]);
    ldsB[kb][nt][ln] = frag;
  }
  __syncthreads();

  const int wave   = tid >> 5;
  const int lane   = tid & 31;
  const int tile   = blockIdx.x * GEMM_WAVES + wave;
  const int ntiles = (n + 15) >> 4;
  if (tile >= ntiles) return;  // wave-uniform; surviving waves keep EXEC all-1s

  const int arow0 = tile * 16 + (lane & 15);
  const int arow  = arow0 < n ? arow0 : (n - 1);     // clamp for tail tile loads
  const int koff  = (lane >> 4) * 16;

  v8f acc0 = {}, acc1 = {}, acc2 = {}, acc3 = {};

#pragma unroll
  for (int kb = 0; kb < NKB; ++kb) {
    const float* ap = A + arow * K + kb * 32 + koff;
    const float4 f0 = *(const float4*)(ap);
    const float4 f1 = *(const float4*)(ap + 4);
    const float4 f2 = *(const float4*)(ap + 8);
    const float4 f3 = *(const float4*)(ap + 12);
    v16bf af;
    af[0]  = f2bf(f0.x); af[1]  = f2bf(f0.y); af[2]  = f2bf(f0.z); af[3]  = f2bf(f0.w);
    af[4]  = f2bf(f1.x); af[5]  = f2bf(f1.y); af[6]  = f2bf(f1.z); af[7]  = f2bf(f1.w);
    af[8]  = f2bf(f2.x); af[9]  = f2bf(f2.y); af[10] = f2bf(f2.z); af[11] = f2bf(f2.w);
    af[12] = f2bf(f3.x); af[13] = f2bf(f3.y); af[14] = f2bf(f3.z); af[15] = f2bf(f3.w);

    acc0 = __builtin_amdgcn_wmma_f32_16x16x32_bf16(false, af, false, ldsB[kb][0][lane],
                                                   (short)0, acc0, false, false);
    acc1 = __builtin_amdgcn_wmma_f32_16x16x32_bf16(false, af, false, ldsB[kb][1][lane],
                                                   (short)0, acc1, false, false);
    acc2 = __builtin_amdgcn_wmma_f32_16x16x32_bf16(false, af, false, ldsB[kb][2][lane],
                                                   (short)0, acc2, false, false);
    acc3 = __builtin_amdgcn_wmma_f32_16x16x32_bf16(false, af, false, ldsB[kb][3][lane],
                                                   (short)0, acc3, false, false);
  }

  // C/D layout: VGPR r -> row tile*16 + r + (lane>=16 ? 8 : 0), col = lane&15 (+16*nt)
  const int rbase = tile * 16 + (lane >> 4) * 8;
  const int cbase = lane & 15;
#pragma unroll
  for (int r = 0; r < 8; ++r) {
    const int row = rbase + r;
    if (row < n) {
      float* cp = C + row * 64 + cbase;
      cp[0]  = acc0[r];
      cp[16] = acc1[r];
      cp[32] = acc2[r];
      cp[48] = acc3[r];
    }
  }
}

// ---------------------------------------------------------------------------
// out[v][c] = bias[c] + xw[v][c] * dinv[v]^2   (self-loop term fused with init)
// ---------------------------------------------------------------------------
__global__ void k_init_agg(const float* __restrict__ xw, const float* __restrict__ dinv,
                           const float* __restrict__ bias, float* __restrict__ out, int n) {
  int idx = blockIdx.x * blockDim.x + threadIdx.x;
  if (idx >= n * 64) return;
  const int v = idx >> 6, c = idx & 63;
  const float di = dinv[v];
  out[idx] = bias[c] + xw[idx] * di * di;
}

// ---------------------------------------------------------------------------
// Edge scatter: one wave per edge, one lane per 2 channels.
// out[dst] += xw[src] * dinv[src]*dinv[dst]   (hardware global_atomic_add_f32)
// ---------------------------------------------------------------------------
__global__ void k_scatter(const int* __restrict__ src, const int* __restrict__ dst,
                          const float* __restrict__ dinv, const float* __restrict__ xw,
                          float* __restrict__ out, int ne) {
  const int wid  = (blockIdx.x * blockDim.x + threadIdx.x) >> 5;
  const int lane = threadIdx.x & 31;
  if (wid >= ne) return;
  const int s = src[wid], d = dst[wid];
  const float norm = dinv[s] * dinv[d];
  const float2 m = ((const float2*)(xw + s * 64))[lane];
  float* op = out + d * 64 + lane * 2;
  unsafeAtomicAdd(op + 0, m.x * norm);
  unsafeAtomicAdd(op + 1, m.y * norm);
}

// ---------------------------------------------------------------------------
// BatchNorm statistics: per-channel sum / sumsq over n rows.
// 256 threads = 4 row-groups x 64 channels; fully coalesced 1KB row reads.
// ---------------------------------------------------------------------------
__global__ void k_bn_stats(const float* __restrict__ h, float* __restrict__ stats, int n) {
  __shared__ float s0[256], s1[256];
  const int tid = threadIdx.x;
  const int c = tid & 63, rg = tid >> 6;
  float sum = 0.f, sq = 0.f;
  for (int v = blockIdx.x * 4 + rg; v < n; v += gridDim.x * 4) {
    const float x = h[v * 64 + c];
    sum += x;
    sq  += x * x;
  }
  s0[tid] = sum;
  s1[tid] = sq;
  __syncthreads();
  if (tid < 64) {
    sum = s0[tid] + s0[tid + 64] + s0[tid + 128] + s0[tid + 192];
    sq  = s1[tid] + s1[tid + 64] + s1[tid + 128] + s1[tid + 192];
    unsafeAtomicAdd(&stats[tid], sum);
    unsafeAtomicAdd(&stats[64 + tid], sq);
  }
}

__global__ void k_bn_finalize(float* __restrict__ stats, const float* __restrict__ gamma,
                              const float* __restrict__ beta, int n) {
  const int c = threadIdx.x;
  if (c >= 64) return;
  const float inv_n = 1.0f / (float)n;
  const float mu  = stats[c] * inv_n;
  const float var = stats[64 + c] * inv_n - mu * mu;
  const float rs  = rsqrtf(var + 1e-5f);
  const float scale = gamma[c] * rs;
  stats[128 + c] = scale;
  stats[192 + c] = beta[c] - mu * scale;
}

__global__ void k_bn_relu(float* __restrict__ h, const float* __restrict__ stats, int n) {
  int idx = blockIdx.x * blockDim.x + threadIdx.x;
  if (idx >= n * 64) return;
  const int c = idx & 63;
  h[idx] = fmaxf(h[idx] * stats[128 + c] + stats[192 + c], 0.0f);
}

// ---------------------------------------------------------------------------
// Global mean pool: atomic segment-sum by (sorted) batch id + node counts.
// ---------------------------------------------------------------------------
__global__ void k_pool(const float* __restrict__ h, const int* __restrict__ batch,
                       float* __restrict__ pooled, float* __restrict__ cnt, int n) {
  int idx = blockIdx.x * blockDim.x + threadIdx.x;
  if (idx >= n * 64) return;
  const int v = idx >> 6, c = idx & 63;
  const int g = batch[v];
  unsafeAtomicAdd(&pooled[g * 64 + c], h[idx]);
  if (c == 0) unsafeAtomicAdd(&cnt[g], 1.0f);
}

__global__ void k_pool_div(float* __restrict__ pooled, const float* __restrict__ cnt, int ngraphs) {
  int idx = blockIdx.x * blockDim.x + threadIdx.x;
  if (idx >= ngraphs * 64) return;
  pooled[idx] /= fmaxf(cnt[idx >> 6], 1.0f);
}

// ---------------------------------------------------------------------------
// MLP head: [256 x 128] -> 64 -> 32 -> 2.  2.6 MFLOP: one 256-thread block.
// ---------------------------------------------------------------------------
__global__ void k_mlp(const float* __restrict__ pb, const float* __restrict__ pt,
                      const float* __restrict__ W1, const float* __restrict__ b1,
                      const float* __restrict__ W2, const float* __restrict__ b2,
                      const float* __restrict__ W3, const float* __restrict__ b3,
                      float* __restrict__ out, int ngraphs) {
  const int g = threadIdx.x;
  if (g >= ngraphs) return;
  float h1[64];
  for (int o = 0; o < 64; ++o) {
    float acc = b1[o];
    for (int i = 0; i < 64; ++i) acc += pb[g * 64 + i] * W1[i * 64 + o];
    for (int i = 0; i < 64; ++i) acc += pt[g * 64 + i] * W1[(64 + i) * 64 + o];
    h1[o] = fmaxf(acc, 0.0f);
  }
  float h2[32];
  for (int o = 0; o < 32; ++o) {
    float acc = b2[o];
    for (int i = 0; i < 64; ++i) acc += h1[i] * W2[i * 32 + o];
    h2[o] = fmaxf(acc, 0.0f);
  }
  for (int o = 0; o < 2; ++o) {
    float acc = b3[o];
    for (int i = 0; i < 32; ++i) acc += h2[i] * W3[i * 2 + o];
    out[g * 2 + o] = acc;
  }
}

// ---------------------------------------------------------------------------
// Launch
// ---------------------------------------------------------------------------
extern "C" void kernel_launch(void* const* d_in, const int* in_sizes, int n_in,
                              void* d_out, int out_size, void* d_ws, size_t ws_size,
                              hipStream_t stream) {
  (void)n_in; (void)out_size; (void)ws_size;
  const int n  = in_sizes[0] / 128;  // 100000 nodes
  const int ne = in_sizes[1] / 2;    // 1200000 edges
  const int G  = 256;

  const float* block_x   = (const float*)d_in[0];
  const int*   block_ei  = (const int*)d_in[1];
  const int*   block_bat = (const int*)d_in[2];
  const float* tx_x      = (const float*)d_in[3];
  const int*   tx_ei     = (const int*)d_in[4];
  const int*   tx_bat    = (const int*)d_in[5];
  // params flattened in dict insertion order
  const float* p[22];
  for (int i = 0; i < 22; ++i) p[i] = (const float*)d_in[6 + i];
  const float* fc1W = p[16]; const float* fc1b = p[17];
  const float* fc2W = p[18]; const float* fc2b = p[19];
  const float* fc3W = p[20]; const float* fc3b = p[21];

  // workspace layout (floats)
  float* ws       = (float*)d_ws;
  float* bufA     = ws;                                  // n*64
  float* bufB     = bufA + (size_t)n * 64;               // n*64
  float* dinv     = bufB + (size_t)n * 64;               // n
  float* stats    = dinv + n;                            // 256
  float* pooled_b = stats + 256;                         // G*64
  float* pooled_t = pooled_b + (size_t)G * 64;           // G*64
  float* cnt      = pooled_t + (size_t)G * 64;           // G

  const int TB        = 256;
  const int nElemBlks = (n * 64 + TB - 1) / TB;
  const int tiles     = (n + 15) / 16;
  const int gemmBlks  = (tiles + GEMM_WAVES - 1) / GEMM_WAVES;
  const int scatBlks  = (ne + 7) / 8;       // one wave per edge

  auto encoder = [&](const float* x, const int* ei, const int* batch,
                     const float* W1, const float* b1, const float* W2, const float* b2,
                     const float* g1, const float* be1, const float* g2, const float* be2,
                     float* pooled) {
    const int* srcv = ei;
    const int* dstv = ei + ne;

    // symmetric GCN norm
    k_fill<<<(n + TB - 1) / TB, TB, 0, stream>>>(dinv, n, 1.0f);
    k_count_deg<<<(ne + TB - 1) / TB, TB, 0, stream>>>(dstv, dinv, ne);
    k_rsqrt<<<(n + TB - 1) / TB, TB, 0, stream>>>(dinv, n);

    // layer 1: xw = x @ W1 ; agg = b1 + self-loop + edge scatter ; BN ; relu
    k_gemm_bf16<128><<<gemmBlks, TB, 0, stream>>>(x, W1, bufA, n);
    k_init_agg<<<nElemBlks, TB, 0, stream>>>(bufA, dinv, b1, bufB, n);
    k_scatter<<<scatBlks, TB, 0, stream>>>(srcv, dstv, dinv, bufA, bufB, ne);
    k_fill<<<1, 128, 0, stream>>>(stats, 128, 0.0f);
    k_bn_stats<<<512, TB, 0, stream>>>(bufB, stats, n);
    k_bn_finalize<<<1, 64, 0, stream>>>(stats, g1, be1, n);
    k_bn_relu<<<nElemBlks, TB, 0, stream>>>(bufB, stats, n);

    // layer 2
    k_gemm_bf16<64><<<gemmBlks, TB, 0, stream>>>(bufB, W2, bufA, n);
    k_init_agg<<<nElemBlks, TB, 0, stream>>>(bufA, dinv, b2, bufB, n);
    k_scatter<<<scatBlks, TB, 0, stream>>>(srcv, dstv, dinv, bufA, bufB, ne);
    k_fill<<<1, 128, 0, stream>>>(stats, 128, 0.0f);
    k_bn_stats<<<512, TB, 0, stream>>>(bufB, stats, n);
    k_bn_finalize<<<1, 64, 0, stream>>>(stats, g2, be2, n);
    k_bn_relu<<<nElemBlks, TB, 0, stream>>>(bufB, stats, n);

    // mean pool
    k_fill<<<(G * 64 + TB - 1) / TB, TB, 0, stream>>>(pooled, G * 64, 0.0f);
    k_fill<<<1, TB, 0, stream>>>(cnt, G, 0.0f);
    k_pool<<<nElemBlks, TB, 0, stream>>>(bufB, batch, pooled, cnt, n);
    k_pool_div<<<(G * 64 + TB - 1) / TB, TB, 0, stream>>>(pooled, cnt, G);
  };

  encoder(block_x, block_ei, block_bat,
          p[0], p[1], p[2], p[3], p[4], p[5], p[6], p[7], pooled_b);
  encoder(tx_x, tx_ei, tx_bat,
          p[8], p[9], p[10], p[11], p[12], p[13], p[14], p[15], pooled_t);

  k_mlp<<<1, TB, 0, stream>>>(pooled_b, pooled_t, fc1W, fc1b, fc2W, fc2b, fc3W, fc3b,
                              (float*)d_out, G);
}
